// SO3Convolution_64931315581332
// MI455X (gfx1250) — matrix operand order repaired and checked by reading, not compiled, not run on hardware
//
#include <hip/hip_runtime.h>

#define NT 5456
#define PI_D 3.14159265358979323846

typedef _Float16 v16h __attribute__((ext_vector_type(16)));
typedef _Float16 h8   __attribute__((ext_vector_type(8)));
typedef float    v8f  __attribute__((ext_vector_type(8)));

union frag16 { v16h v; h8 h[2]; };

__device__ __forceinline__ int lofs_f(int l){ return (4*l*l*l - l)/3; }

__device__ __forceinline__ int decode_l(int e){
  int l = 0;
  while (lofs_f(l+1) <= e) ++l;
  return l;
}

__device__ __forceinline__ void build_lf(double* lf){
  lf[0] = 0.0;
  for (int i = 1; i < 32; ++i) lf[i] = lf[i-1] + log((double)i);
}

__device__ double wig_d(int l, int mp, int m, double beta, const double* lf){
  double cb = cos(0.5*beta), sb = sin(0.5*beta);
  double pref = 0.5*(lf[l+mp] + lf[l-mp] + lf[l+m] + lf[l-m]);
  int s0 = (m-mp) > 0 ? (m-mp) : 0;
  int s1 = (l+m) < (l-mp) ? (l+m) : (l-mp);
  double acc = 0.0;
  for (int s = s0; s <= s1; ++s){
    double logden = lf[l+m-s] + lf[s] + lf[mp-m+s] + lf[l-mp-s];
    double t = exp(pref - logden)
             * pow(cb, (double)(2*l + m - mp - 2*s))
             * pow(sb, (double)(mp - m + 2*s));
    acc += ((mp - m + s) & 1) ? -t : t;
  }
  return acc;
}

// ---------------- table kernels ----------------

// BN[n][2*mh+c]: forward DFT twiddle e^{-2pi i m~ n /64} = (cos, -sin); cols 62,63 = 0
// CI[mh][y][c]: inverse twiddle e^{+2pi i m~ y /32} = (cos, +sin)
__global__ __launch_bounds__(256) void k_misc(float* BN, float* CI){
  int t = blockIdx.x*256 + threadIdx.x;
  if (t < 4096){
    int n = t >> 6, col = t & 63;
    int mh = col >> 1, c = col & 1;
    float v = 0.f;
    if (mh < 31){
      double th = 2.0*PI_D*(double)((mh-15)*n)/64.0;
      v = (float)(c ? -sin(th) : cos(th));
    }
    BN[t] = v;
  } else if (t < 6080){
    int u = t - 4096;
    int mh = u >> 6, r = u & 63, y = r >> 1, c = r & 1;
    double th = 2.0*PI_D*(double)((mh-15)*y)/32.0;
    CI[u] = (float)(c ? sin(th) : cos(th));
  }
}

// WF[j][entry] = wigner_d(l, mp, m, beta_in_j) * w_j / 8192
__global__ __launch_bounds__(256) void k_wf(float* WF){
  int t = blockIdx.x*256 + threadIdx.x;
  if (t >= 64*NT) return;
  int e = t % NT, j = t / NT;
  int l = decode_l(e);
  int w = 2*l+1, le = e - lofs_f(l);
  int mp = le / w - l, m = le % w - l;
  double lf[32]; build_lf(lf);
  double beta = ((double)j + 0.5) * PI_D / 64.0;
  double jj = (double)(2*j + 1);
  double wsum = 0.0;
  for (int k = 0; k < 32; ++k)
    wsum += sin(PI_D*jj*(double)(2*k+1)/128.0) / (double)(2*k+1);
  double wj = (2.0/32.0) * sin(PI_D*jj/128.0) * wsum;
  double d = wig_d(l, mp, m, beta, lf);
  WF[(size_t)j*NT + e] = (float)(d * wj / 8192.0);
}

// WI[jo][entry] = (2l+1) * wigner_d(l, mp, m, beta_out_jo)
__global__ __launch_bounds__(256) void k_wi(float* WI){
  int t = blockIdx.x*256 + threadIdx.x;
  if (t >= 32*NT) return;
  int e = t % NT, jo = t / NT;
  int l = decode_l(e);
  int w = 2*l+1, le = e - lofs_f(l);
  int mp = le / w - l, m = le % w - l;
  double lf[32]; build_lf(lf);
  double beta = ((double)jo + 0.5) * PI_D / 32.0;
  WI[(size_t)jo*NT + e] = (float)((double)w * wig_d(l, mp, m, beta, lf));
}

// DG[g][entry] = wigner_d(l, mp, m, beta_g)
__global__ __launch_bounds__(256) void k_dg(float* DG){
  int t = blockIdx.x*256 + threadIdx.x;
  if (t >= 24*NT) return;
  int e = t % NT, g = t / NT;
  int l = decode_l(e);
  int w = 2*l+1, le = e - lofs_f(l);
  int mp = le / w - l, m = le % w - l;
  double lf[32]; build_lf(lf);
  double beta = (double)((g >> 3) + 1) * PI_D / 24.0;
  DG[(size_t)g*NT + e] = (float)wig_d(l, mp, m, beta, lf);
}

// FY[entry(n,k)][i][o] (complex) = sum_g ker[i,o,g]*SCALING*d_g * e^{+i alpha_g n~}
__global__ __launch_bounds__(256) void k_fy(const float* __restrict__ DG,
                                            const float* __restrict__ ker,
                                            float* __restrict__ FY){
  int t = blockIdx.x*256 + threadIdx.x;
  if (t >= 32*NT) return;
  int e = t % NT, o = t / NT;
  int l = decode_l(e);
  int w = 2*l+1, le = e - lofs_f(l);
  int nt = le / w - l;  // n~ (signed)
  const float S = 0.14433756729740643f;  // 1/sqrt(48)
  float accr[16], acci[16];
  #pragma unroll
  for (int i = 0; i < 16; ++i){ accr[i] = 0.f; acci[i] = 0.f; }
  for (int g = 0; g < 24; ++g){
    float d = DG[(size_t)g*NT + e];
    float alpha = (float)(g & 7) * 0.78539816339744831f;
    float sa, ca;
    sincosf(alpha * (float)nt, &sa, &ca);
    float wr = S*d*ca, wim = S*d*sa;
    #pragma unroll
    for (int i = 0; i < 16; ++i){
      float kv = ker[(i*32 + o)*24 + g];
      accr[i] += kv*wr; acci[i] += kv*wim;
    }
  }
  #pragma unroll
  for (int i = 0; i < 16; ++i){
    size_t fi = ((size_t)e*16 + i)*64 + (size_t)o*2;
    FY[fi] = accr[i]; FY[fi+1] = acci[i];
  }
}

// ---------------- main WMMA DFT kernel ----------------
// One block per (b,f). Per j-slab: partial DFT2 (64x64 -> 31x31 complex) via
// v_wmma_f32_16x16x32_f16 with contiguous (b128) LDS fragment loads, then
// Wigner-forward weighted accumulation into an LDS Fx accumulator.
__global__ __launch_bounds__(256) void k_dft(const float* __restrict__ xg,
                                             const float* __restrict__ BN,
                                             const float* __restrict__ WF,
                                             float* __restrict__ FX){
  __shared__ _Float16 sBt[64][64];   // twiddle transposed: [col=2m+c][n]
  __shared__ _Float16 sA[64][64];    // input slab, row-major [row][n]
  __shared__ _Float16 sX1t[64][64];  // stage-n out transposed: [col=2n+c][row]
  __shared__ float    sPt[64][32];   // Er*X1 transposed [col][m^] (then X2)
  __shared__ float    sQt[64][32];   // Ei*X1 transposed
  __shared__ float    sFx[2*NT];

  int bf = blockIdx.x;
  int tid = threadIdx.x;
  int lane = tid & 31, wv = tid >> 5;
  int ln = lane & 15, half = lane >> 4;

  for (int t = tid; t < 4096; t += 256){
    int n = t >> 6, col = t & 63;
    sBt[col][n] = (_Float16)BN[t];
  }
  for (int t = tid; t < 2*NT; t += 256) sFx[t] = 0.f;
  __syncthreads();

  for (int j = 0; j < 64; ++j){
    // prefetch next slab (global_prefetch_b8)
    if (j + 1 < 64 && tid < 128)
      __builtin_prefetch(xg + (((size_t)bf*64 + j + 1) * 4096) + (size_t)tid*32, 0, 1);

    // load slab j as f16 (row-major)
    const float4* src = (const float4*)(xg + (((size_t)bf*64 + j) * 4096));
    for (int t = tid; t < 1024; t += 256){
      float4 v = src[t];
      _Float16* pa = &((_Float16*)sA)[t*4];
      pa[0] = (_Float16)v.x; pa[1] = (_Float16)v.y;
      pa[2] = (_Float16)v.z; pa[3] = (_Float16)v.w;
    }
    __syncthreads();

    // ---- stage n: X1 = slab(64x64) * BN(64x62pad64) ----
    #pragma unroll
    for (int rep = 0; rep < 2; ++rep){
      int t8 = wv*2 + rep;
      int mt = t8 >> 2, ntile = t8 & 3;
      v8f c = {0.f,0.f,0.f,0.f,0.f,0.f,0.f,0.f};
      #pragma unroll
      for (int ks = 0; ks < 2; ++ks){
        frag16 a, b;
        int m = mt*16 + ln;
        a.h[0] = *(const h8*)&sA[m][ks*32 + half*8];
        a.h[1] = *(const h8*)&sA[m][ks*32 + 16 + half*8];
        b.h[0] = *(const h8*)&sBt[ntile*16 + ln][ks*32 + half*8];
        b.h[1] = *(const h8*)&sBt[ntile*16 + ln][ks*32 + 16 + half*8];
        c = __builtin_amdgcn_wmma_f32_16x16x32_f16(false, a.v, false, b.v, (short)0, c, false, false);
      }
      // C tile -> sX1t[col][row]: contiguous over r
      h8 hv;
      #pragma unroll
      for (int r = 0; r < 8; ++r) hv[r] = (_Float16)c[r];
      *(h8*)&sX1t[ntile*16 + ln][mt*16 + half*8] = hv;
    }
    __syncthreads();

    // ---- stage m: P = Er*X1, Q = Ei*X1 (A rows = sBt rows 2m+pq) ----
    #pragma unroll
    for (int rep = 0; rep < 2; ++rep){
      int t16 = wv*2 + rep;
      int pq = t16 & 1;
      int tt = t16 >> 1;
      int mt = tt >> 2, ntile = tt & 3;
      v8f c = {0.f,0.f,0.f,0.f,0.f,0.f,0.f,0.f};
      #pragma unroll
      for (int ks = 0; ks < 2; ++ks){
        frag16 a, b;
        int mm = mt*16 + ln;
        a.h[0] = *(const h8*)&sBt[2*mm + pq][ks*32 + half*8];
        a.h[1] = *(const h8*)&sBt[2*mm + pq][ks*32 + 16 + half*8];
        b.h[0] = *(const h8*)&sX1t[ntile*16 + ln][ks*32 + half*8];
        b.h[1] = *(const h8*)&sX1t[ntile*16 + ln][ks*32 + 16 + half*8];
        c = __builtin_amdgcn_wmma_f32_16x16x32_f16(false, a.v, false, b.v, (short)0, c, false, false);
      }
      float (*dst)[32] = pq ? sQt : sPt;
      float4 f0 = {c[0], c[1], c[2], c[3]};
      float4 f1 = {c[4], c[5], c[6], c[7]};
      *(float4*)&dst[ntile*16 + ln][mt*16 + half*8]     = f0;
      *(float4*)&dst[ntile*16 + ln][mt*16 + half*8 + 4] = f1;
    }
    __syncthreads();

    // ---- combine complex: X2 (into sPt in-place) ----
    for (int cell = tid; cell < 961; cell += 256){
      int mh = cell / 31, nh = cell % 31;
      float p0 = sPt[2*nh][mh], p1 = sPt[2*nh+1][mh];
      float q0 = sQt[2*nh][mh], q1 = sQt[2*nh+1][mh];
      sPt[2*nh][mh]   = p0 - q1;
      sPt[2*nh+1][mh] = p1 + q0;
    }
    __syncthreads();

    // ---- Wigner-forward accumulation over j ----
    const float* wfj = WF + (size_t)j*NT;
    for (int l = 0; l < 16; ++l){
      int w = 2*l+1, base = lofs_f(l), sz = w*w;
      for (int e = tid; e < sz; e += 256){
        int mh = e / w - l + 15;
        int nh = e % w - l + 15;
        float wf = wfj[base + e];
        sFx[2*(base+e)]   += wf * sPt[2*nh][mh];
        sFx[2*(base+e)+1] += wf * sPt[2*nh+1][mh];
      }
    }
    __syncthreads();
  }

  float* dst = FX + (size_t)bf*2*NT;
  for (int t = tid; t < 2*NT; t += 256) dst[t] = sFx[t];
}

// ---------------- Fz: per (l,m) complex GEMM ----------------
__global__ __launch_bounds__(256) void k_fz(const float* __restrict__ FX,
                                            const float* __restrict__ FY,
                                            float* __restrict__ FZ){
  int v = blockIdx.x;
  int l = 0;
  while ((l+1)*(l+1) <= v) ++l;
  int mh = v - l*l;
  int w = 2*l+1, base = lofs_f(l);
  int cnt = 32*w;
  for (int t = threadIdx.x; t < cnt; t += 256){
    int o = t / w, nh = t % w;
    float ar[8], ai[8];
    #pragma unroll
    for (int b = 0; b < 8; ++b){ ar[b] = 0.f; ai[b] = 0.f; }
    for (int k = 0; k < w; ++k){
      for (int i = 0; i < 16; ++i){
        size_t fyi = ((size_t)(base + nh*w + k)*16 + i)*64 + (size_t)o*2;
        float yr = FY[fyi], yi = FY[fyi+1];
        size_t fxi = ((size_t)i*NT + base + mh*w + k)*2;
        #pragma unroll
        for (int b = 0; b < 8; ++b){
          size_t off = fxi + (size_t)b*16*NT*2;
          float xr = FX[off], xi = FX[off+1];
          ar[b] += xr*yr - xi*yi;
          ai[b] += xr*yi + xi*yr;
        }
      }
    }
    #pragma unroll
    for (int b = 0; b < 8; ++b){
      size_t zi = ((size_t)(b*32 + o)*NT + base + mh*w + nh)*2;
      FZ[zi] = ar[b]; FZ[zi+1] = ai[b];
    }
  }
}

// ---------------- WIG_INV + inverse DFT (WMMA) + bias ----------------
// Per (b,o,jo): S(31x31 cplx) = sum_l WI*Fz ; then out = Re(IDFT2(S)) via
// P=cos*S, Q=sin*S (K=32), combine -> T, then T*B2 (K=64), all f16 WMMA.
__global__ __launch_bounds__(256) void k_out(const float* __restrict__ FZ,
                                             const float* __restrict__ WI,
                                             const float* __restrict__ CI,
                                             const float* __restrict__ bias,
                                             float* __restrict__ out){
  __shared__ _Float16 sSt[64][32];  // S transposed [col=2n+c][m^], zero-padded
  __shared__ _Float16 sAr[32][32];  // cos twiddle [y][m^] (col 31 = 0)
  __shared__ _Float16 sAi[32][32];  // sin twiddle [y][m^]
  __shared__ _Float16 sB2[32][64];  // step3 B transposed [x][2n+c] (rows 62,63=0)
  __shared__ float    sPt[64][32];  // P transposed [col][y]
  __shared__ float    sQt[64][32];
  __shared__ _Float16 sT[32][64];   // T row-major [y][2n+c] (cols 62,63=0)

  int id = blockIdx.x;
  int b = id >> 10, o = (id >> 5) & 31, jo = id & 31;
  int tid = threadIdx.x, lane = tid & 31, wv = tid >> 5;
  int ln = lane & 15, half = lane >> 4;

  for (int t = tid; t < 1024; t += 256){
    int y = t >> 5, m = t & 31;
    float cr = 0.f, ci = 0.f;
    if (m < 31){ cr = CI[(m*32 + y)*2]; ci = CI[(m*32 + y)*2 + 1]; }
    sAr[y][m] = (_Float16)cr;
    sAi[y][m] = (_Float16)ci;
  }
  for (int t = tid; t < 2048; t += 256){
    int x = t >> 6, kc = t & 63;
    int n = kc >> 1, c = kc & 1;
    float v = 0.f;
    if (n < 31) v = c ? -CI[(n*32 + x)*2 + 1] : CI[(n*32 + x)*2];
    sB2[x][kc] = (_Float16)v;
    ((_Float16*)sSt)[t] = (_Float16)0.f;
  }
  __syncthreads();

  // step 1: build S (f16, transposed)
  const float* fz = FZ + (size_t)(b*32 + o)*NT*2;
  const float* wi = WI + (size_t)jo*NT;
  for (int cell = tid; cell < 961; cell += 256){
    int mh = cell / 31, nh = cell % 31;
    int mt = mh - 15, ns = nh - 15;
    int am = mt < 0 ? -mt : mt, an = ns < 0 ? -ns : ns;
    int lmin = am > an ? am : an;
    float ar = 0.f, ai = 0.f;
    for (int l = lmin; l < 16; ++l){
      int idx = lofs_f(l) + (mt + l)*(2*l+1) + (ns + l);
      float w = wi[idx];
      ar += w * fz[2*idx];
      ai += w * fz[2*idx+1];
    }
    sSt[2*nh][mh]   = (_Float16)ar;
    sSt[2*nh+1][mh] = (_Float16)ai;
  }
  __syncthreads();

  // step 2: P = cos*S, Q = sin*S  (M=32, K=32, N=64)
  #pragma unroll
  for (int rep = 0; rep < 2; ++rep){
    int t16 = wv*2 + rep;
    int pq = t16 & 1, tt = t16 >> 1;
    int mt = tt >> 2, ntile = tt & 3;
    frag16 a, bb;
    const _Float16 (*Am)[32] = pq ? sAi : sAr;
    int y = mt*16 + ln;
    a.h[0]  = *(const h8*)&Am[y][half*8];
    a.h[1]  = *(const h8*)&Am[y][16 + half*8];
    bb.h[0] = *(const h8*)&sSt[ntile*16 + ln][half*8];
    bb.h[1] = *(const h8*)&sSt[ntile*16 + ln][16 + half*8];
    v8f c = {0.f,0.f,0.f,0.f,0.f,0.f,0.f,0.f};
    c = __builtin_amdgcn_wmma_f32_16x16x32_f16(false, a.v, false, bb.v, (short)0, c, false, false);
    float (*dst)[32] = pq ? sQt : sPt;
    float4 f0 = {c[0], c[1], c[2], c[3]};
    float4 f1 = {c[4], c[5], c[6], c[7]};
    *(float4*)&dst[ntile*16 + ln][mt*16 + half*8]     = f0;
    *(float4*)&dst[ntile*16 + ln][mt*16 + half*8 + 4] = f1;
  }
  __syncthreads();

  // combine into T (f16, row-major)
  for (int t = tid; t < 1024; t += 256){
    int y = t >> 5, n = t & 31;
    if (n < 31){
      float tre = sPt[2*n][y]   - sQt[2*n+1][y];
      float tim = sPt[2*n+1][y] + sQt[2*n][y];
      sT[y][2*n]   = (_Float16)tre;
      sT[y][2*n+1] = (_Float16)tim;
    } else {
      sT[y][62] = (_Float16)0.f;
      sT[y][63] = (_Float16)0.f;
    }
  }
  __syncthreads();

  // step 3: out = T(32x64) * B2(64x32), waves 0..3
  float bv = bias[o];
  if (wv < 4){
    int mt = wv >> 1, ntile = wv & 1;
    v8f c = {0.f,0.f,0.f,0.f,0.f,0.f,0.f,0.f};
    #pragma unroll
    for (int ks = 0; ks < 2; ++ks){
      frag16 a, bb;
      int y = mt*16 + ln;
      a.h[0]  = *(const h8*)&sT[y][ks*32 + half*8];
      a.h[1]  = *(const h8*)&sT[y][ks*32 + 16 + half*8];
      bb.h[0] = *(const h8*)&sB2[ntile*16 + ln][ks*32 + half*8];
      bb.h[1] = *(const h8*)&sB2[ntile*16 + ln][ks*32 + 16 + half*8];
      c = __builtin_amdgcn_wmma_f32_16x16x32_f16(false, a.v, false, bb.v, (short)0, c, false, false);
    }
    #pragma unroll
    for (int r = 0; r < 8; ++r){
      int y = mt*16 + r + half*8, x = ntile*16 + ln;
      out[(size_t)id*1024 + y*32 + x] = c[r] + bv;
    }
  }
}

// ---------------- host ----------------
extern "C" void kernel_launch(void* const* d_in, const int* in_sizes, int n_in,
                              void* d_out, int out_size, void* d_ws, size_t ws_size,
                              hipStream_t stream) {
  (void)in_sizes; (void)n_in; (void)out_size; (void)ws_size;
  const float* x    = (const float*)d_in[0];
  const float* ker  = (const float*)d_in[1];
  const float* bias = (const float*)d_in[2];
  float* out = (float*)d_out;

  float* ws = (float*)d_ws;
  float* BN = ws + 0;        // 4096
  float* CI = ws + 4096;     // 1984
  float* WF = ws + 6080;     // 349184
  float* WI = ws + 355264;   // 174592
  float* DG = ws + 529856;   // 130944
  float* FY = ws + 660800;   // 5586944
  float* FX = ws + 6247744;  // 1396736
  float* FZ = ws + 7644480;  // 2793472  (end = 10437952 floats ~ 41.8 MB)

  k_misc<<<24,   256, 0, stream>>>(BN, CI);
  k_wf  <<<1364, 256, 0, stream>>>(WF);
  k_wi  <<<682,  256, 0, stream>>>(WI);
  k_dg  <<<512,  256, 0, stream>>>(DG);
  k_fy  <<<682,  256, 0, stream>>>(DG, ker, FY);
  k_dft <<<128,  256, 0, stream>>>(x, BN, WF, FX);
  k_fz  <<<256,  256, 0, stream>>>(FX, FY, FZ);
  k_out <<<8192, 256, 0, stream>>>(FZ, WI, CI, bias, out);
}
